// EncoderRNN_74783970558462
// MI455X (gfx1250) — matrix operand order repaired and checked by reading, not compiled, not run on hardware
//
#include <hip/hip_runtime.h>
#include <hip/hip_bf16.h>
#include <math.h>

#define T_ 256
#define B_ 64
#define M_ 512
#define N_ 1024

#define XS_LD 520    // 512 + 8-elem pad: rows 16B-aligned, 4-bank skew -> conflict-free
#define HS_LD 1032   // 1024 + 8-elem pad
#define BP    32     // B-panel columns per block (b-split x2)

// ---------------------------------------------------------------- types ----
typedef __attribute__((ext_vector_type(16))) __bf16        v16bf;
typedef __attribute__((ext_vector_type(8)))  float         v8f;
typedef __attribute__((ext_vector_type(8)))  unsigned short ushortx8;

union FragCvt { ushortx8 u[2]; v16bf v; };

__device__ __forceinline__ unsigned short f32_to_bf16_bits(float f) {
  unsigned int u = __float_as_uint(f);
  unsigned int r = 0x7FFFu + ((u >> 16) & 1u);   // round-to-nearest-even
  return (unsigned short)((u + r) >> 16);
}

// A-fragment: 16x32 bf16, M(row)-major source with row stride ld (elements).
// ISA layout: lanes 0-15 -> M=lane, K=0..7 (v0-3), K=16..23 (v4-7);
//             lanes 16-31 -> M=lane-16, K=8..15 (v0-3), K=24..31 (v4-7).
__device__ __forceinline__ v16bf load_frag_a(const unsigned short* base, int ld) {
  const int lane = threadIdx.x & 31;
  const int row  = lane & 15;
  const int k0   = (lane & 16) ? 8 : 0;
  const unsigned short* p = base + (size_t)row * ld + k0;
  FragCvt c;
  c.u[0] = *(const ushortx8*)(p);        // K = k0 .. k0+7
  c.u[1] = *(const ushortx8*)(p + 16);   // K = k0+16 .. k0+23
  return c.v;
}

// B-fragment: 32x16 bf16 (KxN), column-contiguous source with stride ld.
// ISA layout: lanes 0-15 -> N=lane, K=0..7 (v0-3), K=16..23 (v4-7);
//             lanes 16-31 -> N=lane-16, K=8..15 (v0-3), K=24..31 (v4-7).
__device__ __forceinline__ v16bf load_frag_b(const unsigned short* base, int ld) {
  const int lane = threadIdx.x & 31;
  const int col  = lane & 15;
  const int k0   = (lane & 16) ? 8 : 0;
  const unsigned short* p = base + (size_t)col * ld + k0;
  FragCvt c;
  c.u[0] = *(const ushortx8*)(p);        // K = k0 .. k0+7
  c.u[1] = *(const ushortx8*)(p + 16);   // K = k0+16 .. k0+23
  return c.v;
}

// ---------------------------------------------------------- ws layout ------
constexpr size_t OFF_XBF  = 0;                                          // bf16 x   [T][B][M]
constexpr size_t OFF_WT   = OFF_XBF  + (size_t)T_ * B_ * M_ * 2;        // bf16 W^T [2][3][N][M]
constexpr size_t OFF_U    = OFF_WT   + (size_t)6 * N_ * M_ * 2;         // bf16 U   [2][4][N][N]
constexpr size_t OFF_HT   = OFF_U    + (size_t)8 * N_ * N_ * 2;         // bf16 h^T [2 buf][2 dir][B][N]
constexpr size_t OFF_C    = OFF_HT   + (size_t)2 * 2 * B_ * N_ * 2;     // f32  c   [2][N][B]
constexpr size_t OFF_BIAS = OFF_C    + (size_t)2 * N_ * B_ * 4;         // f32  bias[2][4][N]
constexpr size_t WS_NEED  = OFF_BIAS + (size_t)2 * 4 * N_ * 4;          // ~39.0 MiB

// dynamic LDS: x-panel + h-panel + gate-exchange tile
constexpr size_t SMEM_XS    = (size_t)BP * XS_LD * 2;                   // 33280 B
constexpr size_t SMEM_HS    = (size_t)BP * HS_LD * 2;                   // 66048 B
constexpr size_t SMEM_GT    = (size_t)4 * 16 * BP * 4;                  //  8192 B
constexpr size_t SMEM_BYTES = SMEM_XS + SMEM_HS + SMEM_GT;              // 107520 B

// ------------------------------------------------------------- kernels -----
__global__ void cvt_copy_bf16(const float* __restrict__ s, unsigned short* __restrict__ d, int n) {
  int i = blockIdx.x * blockDim.x + threadIdx.x;
  if (i < n) d[i] = f32_to_bf16_bits(s[i]);
}

__global__ void copy_f32(const float* __restrict__ s, float* __restrict__ d, int n) {
  int i = blockIdx.x * blockDim.x + threadIdx.x;
  if (i < n) d[i] = s[i];
}

__global__ void zero_f32(float* __restrict__ d, int n) {
  int i = blockIdx.x * blockDim.x + threadIdx.x;
  if (i < n) d[i] = 0.0f;
}

// Transpose-convert: src f32 [rows][cols] -> dst bf16 [cols][rows].  LDS-tiled.
__global__ void cvt_transpose_bf16(const float* __restrict__ src,
                                   unsigned short* __restrict__ dst,
                                   int rows, int cols) {
  __shared__ unsigned short tile[32][33];
  const int c0 = blockIdx.x * 32, r0 = blockIdx.y * 32;
  const int tx = threadIdx.x, ty = threadIdx.y;              // block 32x8
  for (int i = ty; i < 32; i += 8) {
    int r = r0 + i, c = c0 + tx;
    tile[i][tx] = (r < rows && c < cols) ? f32_to_bf16_bits(src[(size_t)r * cols + c])
                                         : (unsigned short)0;
  }
  __syncthreads();
  for (int i = ty; i < 32; i += 8) {
    int rr = c0 + i, cc = r0 + tx;                           // dst coords [cols][rows]
    if (rr < cols && cc < rows) dst[(size_t)rr * rows + cc] = tile[tx][i];
  }
}

// ---------------------------------------------------------------------------
// One fused recurrent step (GEMM for all 4 gates + gate math), both dirs.
//   grid:  dim3(64, 2, 2) -> 64 n-strips of 16 rows, 2 b-halves, 2 dirs
//   block: 128 threads    -> 4 waves; wave w computes gate w (f,i,o,c) for
//          the SAME 16(n) x 32(b) strip.
// The shared B operands (x_t panel, h^T panel) are cooperatively staged in
// padded LDS once per block -> K-loop B reads are conflict-free ds_load_b128
// on DScnt while the per-gate A rows stream from global on LOADcnt.
// Gate tiles are exchanged through LDS, then the pointwise LSTM math writes
// c (f32), h^T (bf16 ping-pong) and the output slice.
// ---------------------------------------------------------------------------
__global__ void __launch_bounds__(128)
lstm_step_fused(const unsigned short* __restrict__ xbf,
                const unsigned short* __restrict__ wt,
                const unsigned short* __restrict__ ub,
                const unsigned short* __restrict__ hTin,
                unsigned short* __restrict__ hTout,
                float* __restrict__ cst,
                const float* __restrict__ bias,
                float* __restrict__ out, int t) {
  extern __shared__ unsigned short smem[];
  unsigned short* xs = smem;                          // [BP][XS_LD]
  unsigned short* hs = smem + (size_t)BP * XS_LD;     // [BP][HS_LD]
  float* gtile = (float*)(smem + (size_t)BP * XS_LD + (size_t)BP * HS_LD); // [4][16][BP]

  const int dir  = blockIdx.z;
  const int b0   = blockIdx.y * BP;
  const int gate = threadIdx.x >> 5;     // wave id: 0=f 1=i 2=o 3=c
  const int n0   = blockIdx.x * 16;
  const int ts   = dir ? (T_ - 1 - t) : t;
  // faithful to the reference's 'Wo'->W_f binding: o-gate x-proj uses Wf
  const int wsel = (gate == 1) ? 1 : ((gate == 3) ? 2 : 0);

  const unsigned short* Aw = wt  + ((size_t)dir * 3 + wsel) * (size_t)N_ * M_ + (size_t)n0 * M_;
  const unsigned short* Bx = xbf + (size_t)ts * B_ * M_ + (size_t)b0 * M_;
  const unsigned short* Au = ub  + ((size_t)dir * 4 + gate) * (size_t)N_ * N_ + (size_t)n0 * N_;
  const unsigned short* Bh = hTin + (size_t)dir * B_ * N_ + (size_t)b0 * N_;

  // ---- cooperative B-panel staging (identical data for all 4 gate-waves) --
  for (int idx = threadIdx.x; idx < BP * (M_ / 8); idx += 128) {   // 32*64
    const int row = idx >> 6, ch = (idx & 63) * 8;
    *(ushortx8*)(xs + (size_t)row * XS_LD + ch) = *(const ushortx8*)(Bx + (size_t)row * M_ + ch);
  }
  for (int idx = threadIdx.x; idx < BP * (N_ / 8); idx += 128) {   // 32*128
    const int row = idx >> 7, ch = (idx & 127) * 8;
    *(ushortx8*)(hs + (size_t)row * HS_LD + ch) = *(const ushortx8*)(Bh + (size_t)row * N_ + ch);
  }
  __syncthreads();

  v8f acc0 = {}, acc1 = {};

  // ---- W^T x_t contribution: K over M=512 (A from global, B from LDS) ----
  for (int k = 0; k < M_; k += 32) {
    v16bf a  = load_frag_a(Aw + k, M_);
    v16bf b0f = load_frag_b(xs + k,              XS_LD);
    v16bf b1f = load_frag_b(xs + 16 * XS_LD + k, XS_LD);
    acc0 = __builtin_amdgcn_wmma_f32_16x16x32_bf16(false, a, false, b0f, (short)0, acc0, false, false);
    acc1 = __builtin_amdgcn_wmma_f32_16x16x32_bf16(false, a, false, b1f, (short)0, acc1, false, false);
  }

  // ---- U h contribution: K over N=1024 (A from global, B from LDS) ----
  for (int k = 0; k < N_; k += 32) {
    v16bf a  = load_frag_a(Au + k, N_);
    v16bf b0f = load_frag_b(hs + k,              HS_LD);
    v16bf b1f = load_frag_b(hs + 16 * HS_LD + k, HS_LD);
    acc0 = __builtin_amdgcn_wmma_f32_16x16x32_bf16(false, a, false, b0f, (short)0, acc0, false, false);
    acc1 = __builtin_amdgcn_wmma_f32_16x16x32_bf16(false, a, false, b1f, (short)0, acc1, false, false);
  }

  // ---- park this gate's 16x32 tile in LDS (D layout: lane=col, VGPR=row) --
  {
    const int lane  = threadIdx.x & 31;
    const int col   = lane & 15;
    const int rbase = (lane & 16) ? 8 : 0;
#pragma unroll
    for (int r = 0; r < 8; ++r) {
      gtile[((size_t)gate * 16 + rbase + r) * BP +  0 + col] = acc0[r];
      gtile[((size_t)gate * 16 + rbase + r) * BP + 16 + col] = acc1[r];
    }
  }
  __syncthreads();

  // ---- pointwise LSTM math: 512 elements over 128 threads -----------------
  const float* bb = bias + (size_t)dir * 4 * N_;
  for (int e = threadIdx.x; e < 16 * BP; e += 128) {
    const int nl = e >> 5;               // 0..15
    const int bl = e & 31;               // 0..31
    const int n  = n0 + nl;
    const int b  = b0 + bl;

    const float f  = 1.0f / (1.0f + __expf(-(gtile[(0 * 16 + nl) * BP + bl] + bb[0 * N_ + n])));
    const float i  = 1.0f / (1.0f + __expf(-(gtile[(1 * 16 + nl) * BP + bl] + bb[1 * N_ + n])));
    const float o  = 1.0f / (1.0f + __expf(-(gtile[(2 * 16 + nl) * BP + bl] + bb[2 * N_ + n])));
    const float cb = tanhf(gtile[(3 * 16 + nl) * BP + bl] + bb[3 * N_ + n]);

    const size_t ci = ((size_t)dir * N_ + n) * B_ + b;
    const float cnew = f * cst[ci] + i * cb;
    const float h = o * tanhf(cnew);

    cst[ci] = cnew;
    hTout[((size_t)dir * B_ + b) * N_ + n] = f32_to_bf16_bits(h);
    const int tout = dir ? (T_ - 1 - t) : t;
    out[((size_t)tout * B_ + b) * (2 * N_) + (size_t)dir * N_ + n] = h;
  }
}

// ------------------------------------------------------------- launch ------
extern "C" void kernel_launch(void* const* d_in, const int* in_sizes, int n_in,
                              void* d_out, int out_size, void* d_ws, size_t ws_size,
                              hipStream_t stream) {
  (void)in_sizes; (void)n_in; (void)out_size;
  if (ws_size < WS_NEED) return;

  const float* x      = (const float*)d_in[0];
  const float* hidden = (const float*)d_in[1];
  // per-direction input indices:  W:{+0,+1,+2}  U:{+3..+6}  b:{+7..+10}
  const int base[2] = {2, 13};

  char* ws = (char*)d_ws;
  unsigned short* xbf  = (unsigned short*)(ws + OFF_XBF);
  unsigned short* wt   = (unsigned short*)(ws + OFF_WT);
  unsigned short* ub   = (unsigned short*)(ws + OFF_U);
  unsigned short* hT   = (unsigned short*)(ws + OFF_HT);   // 2 ping-pong bufs
  float*          cst  = (float*)(ws + OFF_C);
  float*          bias = (float*)(ws + OFF_BIAS);
  float*          out  = (float*)d_out;

  const size_t hbuf = (size_t)2 * B_ * N_;                 // elems per buffer

  // ---- one-time-per-call setup (deterministic; ws is poisoned by harness) --
  const int nx = T_ * B_ * M_;
  cvt_copy_bf16<<<(nx + 255) / 256, 256, 0, stream>>>(x, xbf, nx);

  for (int d = 0; d < 2; ++d) {
    for (int g = 0; g < 3; ++g) {   // Wf, Wi, Wc  ->  W^T bf16 [N][M]
      const float* W = (const float*)d_in[base[d] + g];
      cvt_transpose_bf16<<<dim3(N_ / 32, M_ / 32), dim3(32, 8), 0, stream>>>(
          W, wt + ((size_t)d * 3 + g) * (size_t)N_ * M_, M_, N_);
    }
    for (int g = 0; g < 4; ++g) {   // Uf, Ui, Uo, Uc -> bf16 [N][N]
      const float* U = (const float*)d_in[base[d] + 3 + g];
      const int nu = N_ * N_;
      cvt_copy_bf16<<<(nu + 255) / 256, 256, 0, stream>>>(
          U, ub + ((size_t)d * 4 + g) * (size_t)N_ * N_, nu);
    }
    for (int g = 0; g < 4; ++g) {   // bf, bi, bo, bc -> f32 [N]
      const float* bv = (const float*)d_in[base[d] + 7 + g];
      copy_f32<<<(N_ + 255) / 256, 256, 0, stream>>>(
          bv, bias + ((size_t)d * 4 + g) * N_, N_);
    }
    // h0 is [B][N] row-major == exactly the h^T layout the B-fragment wants;
    // initialize ping-pong buffer 0 for both directions.
    const int nh = B_ * N_;
    cvt_copy_bf16<<<(nh + 255) / 256, 256, 0, stream>>>(
        hidden, hT + (size_t)d * B_ * N_, nh);
  }
  zero_f32<<<(2 * N_ * B_ + 255) / 256, 256, 0, stream>>>(cst, 2 * N_ * B_);

  // ---- sequential recurrence: one fused kernel per step, ping-pong h ------
  for (int t = 0; t < T_; ++t) {
    unsigned short* hin  = hT + (size_t)(t & 1) * hbuf;
    unsigned short* hout = hT + (size_t)((t + 1) & 1) * hbuf;
    lstm_step_fused<<<dim3(64, 2, 2), 128, SMEM_BYTES, stream>>>(
        xbf, wt, ub, hin, hout, cst, bias, out, t);
  }
}
